// SlackAwareGAT_78864189489394
// MI455X (gfx1250) — matrix operand order repaired and checked by reading, not compiled, not run on hardware
//
#include <hip/hip_runtime.h>

// ---------------------------------------------------------------------------
// SlackAwareGAT on MI455X (gfx1250, wave32).
//  - Dense 128x128 GEMMs via v_wmma_f32_16x16x32_f16.  Weights are pre-packed
//    into B-fragment register order (one-shot pack kernel) so every fragment
//    load is a pair of ds_load_b128, not 16x ds_load_u16.
//  - Each block computes a 64x128 tile: 4 row-tiles per wave, 16 WMMA/wave,
//    B fragment reused 4x, 4x less weight restaging.
//  - Edge softmax via float-atomic-max (int-punning) + exp/atomic-add.
//  - Aggregation: one wave32 per edge, prefetched+coalesced gather of
//    xp[src], global_atomic_add_f32 scatter into agg[dst]; xp (25.6MB) and
//    the softmax stats live in the 192MB L2.
// ---------------------------------------------------------------------------

typedef __attribute__((ext_vector_type(16))) _Float16 v16h;
typedef __attribute__((ext_vector_type(8)))  _Float16 v8h;
typedef __attribute__((ext_vector_type(8)))  float    v8f;

#define HIDC   128
#define HEADS  4
#define CHPH   32
#define NLAYER 4
#define WPACK_ELEMS (8 * 4 * 32 * 16)   // 16384 halves per packed weight

static __device__ __forceinline__ void atomicMaxF(float* addr, float val) {
  // ordered-int trick; addr must be initialized to -inf
  if (val >= 0.0f) atomicMax((int*)addr, __float_as_int(val));
  else             atomicMin((unsigned int*)addr, __float_as_uint(val));
}

// ---------------------------------------------------------------------------
// fill
// ---------------------------------------------------------------------------
__global__ void fill_f32(float* __restrict__ p, float v, long n) {
  long i = (long)blockIdx.x * blockDim.x + threadIdx.x;
  if (i < n) p[i] = v;
}

// ---------------------------------------------------------------------------
// Pack fp32 weight W[128][128] (k-major) into f16 B-fragment order:
//   Wp[((ct*4+ks)*32 + lane)*16 + t] = W[k][n],
//   k = ks*32 + 16*(lane>>4) + t,  n = ct*16 + (lane&15)
// so each lane's WMMA B fragment is 16 contiguous halves.
// ---------------------------------------------------------------------------
__global__ void pack_weight_kernel(const float* __restrict__ W,
                                   _Float16* __restrict__ Wp) {
  int i = blockIdx.x * blockDim.x + threadIdx.x;
  if (i >= WPACK_ELEMS) return;
  int t    = i & 15;
  int lane = (i >> 4) & 31;
  int ks   = (i >> 9) & 3;
  int ct   = i >> 11;
  int k = ks * 32 + ((lane >> 4) << 4) + t;
  int n = ct * 16 + (lane & 15);
  Wp[i] = (_Float16)W[k * HIDC + n];
}

// ---------------------------------------------------------------------------
// Y[M x 128] = X[M x 128] @ W[128 x 128] (+ bias), f16 WMMA, f32 accumulate.
// Block = 256 threads = 8 waves; block computes a 64x128 tile.  Wave w owns
// output cols [16w,16w+16) for 4 row-tiles.  Packed weight staged as 32KB
// linear copy; fragments are aligned b128 LDS loads.
// ---------------------------------------------------------------------------
__global__ __launch_bounds__(256) void gemm128_wmma_kernel(
    const float* __restrict__ X, const _Float16* __restrict__ Wp,
    const float* __restrict__ bias, float* __restrict__ Y, int nrows)
{
  __shared__ _Float16 Asf[64][HIDC + 8];       // row pitch 136 halves = 272B (16B mult)
  __shared__ _Float16 Wf[WPACK_ELEMS];         // 32KB, fragment order

  const int tid  = threadIdx.x;
  const int row0 = blockIdx.x * 64;

  // stage packed weight: 32KB linear copy as uint4
  {
    const uint4* src = (const uint4*)Wp;
    uint4*       dst = (uint4*)Wf;
    for (int i = tid; i < WPACK_ELEMS * 2 / 16; i += 256) dst[i] = src[i];
  }
  // stage A tile 64x128, fp32 -> f16 (clamp rows for ragged tail)
  for (int i = tid; i < 64 * HIDC; i += 256) {
    int m = i >> 7, k = i & 127;
    int r = row0 + m; if (r >= nrows) r = nrows - 1;
    Asf[m][k] = (_Float16)X[(size_t)r * HIDC + k];
  }
  __syncthreads();

  const int lane = tid & 31;
  const int wave = tid >> 5;          // column tile id 0..7
  const int n0   = wave * 16;
  const int kgrp = lane >> 4;         // lane half (ISA layout)
  const int mrow = lane & 15;
  const int ncol = lane & 15;

  v8f acc[4] = {};
#pragma unroll
  for (int ks = 0; ks < 4; ++ks) {
    // B fragment: 16 contiguous halves per lane (two aligned b128 loads)
    const v8h* bp = (const v8h*)&Wf[((wave * 4 + ks) * 32 + lane) * 16];
    v8h blo = bp[0], bhi = bp[1];
    v16h b = __builtin_shufflevector(blo, bhi,
        0, 1, 2, 3, 4, 5, 6, 7, 8, 9, 10, 11, 12, 13, 14, 15);
#pragma unroll
    for (int rt = 0; rt < 4; ++rt) {
      // A fragment: halves {ks*32+16*kgrp+0..15}
      const _Float16* ar = &Asf[rt * 16 + mrow][ks * 32 + (kgrp << 4)];
      v8h alo = *(const v8h*)ar;
      v8h ahi = *(const v8h*)(ar + 8);
      v16h a = __builtin_shufflevector(alo, ahi,
          0, 1, 2, 3, 4, 5, 6, 7, 8, 9, 10, 11, 12, 13, 14, 15);
      acc[rt] = __builtin_amdgcn_wmma_f32_16x16x32_f16(
          false, a, false, b, (short)0, acc[rt], false, false);
    }
  }

  // C/D layout: VGPR r, lane half -> M = r + 8*half, N = lane&15
#pragma unroll
  for (int rt = 0; rt < 4; ++rt) {
#pragma unroll
    for (int r = 0; r < 8; ++r) {
      int m = row0 + rt * 16 + r + (kgrp << 3);
      int n = n0 + ncol;
      if (m < nrows) {
        float v = acc[rt][r];
        if (bias) v += bias[n];
        Y[(size_t)m * HIDC + n] = v;
      }
    }
  }
}

// ---------------------------------------------------------------------------
// edge bias MLP -> scalar -> atomicAdd crit[dst]   (real edges only)
// ---------------------------------------------------------------------------
__global__ void edge_mlp_kernel(const float* __restrict__ ea,
                                const float* __restrict__ We1, const float* __restrict__ be1,
                                const float* __restrict__ We2, const float* __restrict__ be2,
                                const int* __restrict__ dsts, float* __restrict__ crit, int E)
{
  int e = blockIdx.x * blockDim.x + threadIdx.x;
  if (e >= E) return;
  float a0 = ea[2 * e], a1 = ea[2 * e + 1];
  float acc = 0.f;
#pragma unroll
  for (int j = 0; j < 16; ++j) {
    float t = fmaxf(a0 * We1[j] + a1 * We1[16 + j] + be1[j], 0.f);
    float w = We2[j * 4 + 0] + We2[j * 4 + 1] + We2[j * 4 + 2] + We2[j * 4 + 3];
    acc += t * w;
  }
  float scalar = 0.25f * (acc + be2[0] + be2[1] + be2[2] + be2[3]);
  atomicAdd(&crit[dsts[e]], scalar);
}

// ---------------------------------------------------------------------------
// per-node attention coefficients: as_n[n,h] = <xp[n,h,:], a_src[h,:]>
// ---------------------------------------------------------------------------
__global__ void attn_coef_kernel(const float* __restrict__ xp,
                                 const float* __restrict__ a_s, const float* __restrict__ a_d,
                                 float* __restrict__ as_n, float* __restrict__ ad_n, int Nn)
{
  int i = blockIdx.x * blockDim.x + threadIdx.x;   // n*HEADS + h
  if (i >= Nn * HEADS) return;
  int n = i >> 2, hh = i & 3;
  const float* xr = xp + (size_t)n * HIDC + hh * CHPH;
  float s1 = 0.f, s2 = 0.f;
#pragma unroll 8
  for (int c = 0; c < CHPH; ++c) {
    float v = xr[c];
    s1 += v * a_s[hh * CHPH + c];
    s2 += v * a_d[hh * CHPH + c];
  }
  as_n[i] = s1; ad_n[i] = s2;
}

// ---------------------------------------------------------------------------
// segment-max of leaky_relu(as[src]+ad[dst]) per (dst, head); self loops incl.
// ---------------------------------------------------------------------------
__global__ void edge_max_kernel(const float* __restrict__ as_n, const float* __restrict__ ad_n,
                                const int* __restrict__ srcs, const int* __restrict__ dsts,
                                float* __restrict__ mbuf, int E, int Nn)
{
  long i = (long)blockIdx.x * blockDim.x + threadIdx.x;
  long total = (long)(E + Nn) * HEADS;
  if (i >= total) return;
  int e = (int)(i >> 2), hh = (int)(i & 3);
  int s = (e < E) ? srcs[e] : (e - E);
  int d = (e < E) ? dsts[e] : (e - E);
  float v = as_n[(size_t)s * HEADS + hh] + ad_n[(size_t)d * HEADS + hh];
  v = (v >= 0.f) ? v : 0.2f * v;
  atomicMaxF(&mbuf[(size_t)d * HEADS + hh], v);
}

__global__ void edge_sum_kernel(const float* __restrict__ as_n, const float* __restrict__ ad_n,
                                const int* __restrict__ srcs, const int* __restrict__ dsts,
                                const float* __restrict__ mbuf, float* __restrict__ sbuf,
                                int E, int Nn)
{
  long i = (long)blockIdx.x * blockDim.x + threadIdx.x;
  long total = (long)(E + Nn) * HEADS;
  if (i >= total) return;
  int e = (int)(i >> 2), hh = (int)(i & 3);
  int s = (e < E) ? srcs[e] : (e - E);
  int d = (e < E) ? dsts[e] : (e - E);
  float v = as_n[(size_t)s * HEADS + hh] + ad_n[(size_t)d * HEADS + hh];
  v = (v >= 0.f) ? v : 0.2f * v;
  float p = __expf(v - mbuf[(size_t)d * HEADS + hh]);
  atomicAdd(&sbuf[(size_t)d * HEADS + hh], p);
}

// ---------------------------------------------------------------------------
// aggregation: one wave32 per edge; lane l covers channels {l, l+32, l+64, l+96},
// chunk q belongs to head q.  Prefetched+coalesced xp gather, atomic scatter.
// ---------------------------------------------------------------------------
__global__ __launch_bounds__(256) void aggregate_kernel(
    const float* __restrict__ xp,
    const float* __restrict__ as_n, const float* __restrict__ ad_n,
    const float* __restrict__ mbuf, const float* __restrict__ sbuf,
    const int* __restrict__ srcs, const int* __restrict__ dsts,
    float* __restrict__ agg, int E, int Nn)
{
  int wave = (int)(((long)blockIdx.x * blockDim.x + threadIdx.x) >> 5);
  int lane = threadIdx.x & 31;
  int total = E + Nn;
  if (wave >= total) return;
  int s = (wave < E) ? srcs[wave] : (wave - E);
  int d = (wave < E) ? dsts[wave] : (wave - E);

  const float* xs = xp + (size_t)s * HIDC;
  __builtin_prefetch(xs + lane, 0, 0);   // global_prefetch_b8: warm gather row

  float alpha[HEADS];
#pragma unroll
  for (int hh = 0; hh < HEADS; ++hh) {
    float v = as_n[(size_t)s * HEADS + hh] + ad_n[(size_t)d * HEADS + hh];
    v = (v >= 0.f) ? v : 0.2f * v;
    alpha[hh] = __expf(v - mbuf[(size_t)d * HEADS + hh]) /
                (sbuf[(size_t)d * HEADS + hh] + 1e-16f);
  }
  float* ad = agg + (size_t)d * HIDC;
#pragma unroll
  for (int q = 0; q < HEADS; ++q) {
    int c = q * 32 + lane;
    atomicAdd(&ad[c], xs[c] * alpha[q]);
  }
}

// ---------------------------------------------------------------------------
// h[n] += LN(agg[n]+bl)*g+b + crit[n]     (one node per 128-thread block)
// ---------------------------------------------------------------------------
__global__ __launch_bounds__(128) void post_kernel(
    const float* __restrict__ agg, const float* __restrict__ bl,
    const float* __restrict__ g, const float* __restrict__ b,
    const float* __restrict__ crit, float* __restrict__ h, int Nn)
{
  __shared__ float red[128];
  int n = blockIdx.x, c = threadIdx.x;
  float v = agg[(size_t)n * HIDC + c] + bl[c];
  red[c] = v; __syncthreads();
  for (int s = 64; s > 0; s >>= 1) { if (c < s) red[c] += red[c + s]; __syncthreads(); }
  float mu = red[0] * (1.f / HIDC);
  __syncthreads();
  float dv = v - mu;
  red[c] = dv * dv; __syncthreads();
  for (int s = 64; s > 0; s >>= 1) { if (c < s) red[c] += red[c + s]; __syncthreads(); }
  float inv = rsqrtf(red[0] * (1.f / HIDC) + 1e-5f);
  float out = dv * inv * g[c] + b[c] + crit[n];
  h[(size_t)n * HIDC + c] += out;
}

// ---------------------------------------------------------------------------
// gate = tanh(h@Wg1+bg1)@Wg2+bg2; one wave32 per node, shuffle reduction.
// ---------------------------------------------------------------------------
__global__ __launch_bounds__(256) void gate_kernel(
    const float* __restrict__ h, const float* __restrict__ Wg1,
    const float* __restrict__ bg1, const float* __restrict__ Wg2,
    const float* __restrict__ bg2, const int* __restrict__ batch,
    float* __restrict__ gate, float* __restrict__ gmax, int Nn)
{
  int node = (int)(((long)blockIdx.x * blockDim.x + threadIdx.x) >> 5);
  int lane = threadIdx.x & 31;
  if (node >= Nn) return;
  const float* hr = h + (size_t)node * HIDC;
  float acc = 0.f;
#pragma unroll
  for (int rep = 0; rep < 2; ++rep) {
    int j = lane + rep * 32;                 // hidden unit, HID/2 = 64
    float t = bg1[j];
    for (int c = 0; c < HIDC; ++c) t += hr[c] * Wg1[c * 64 + j];
    acc += tanhf(t) * Wg2[j];
  }
  for (int off = 16; off > 0; off >>= 1) acc += __shfl_xor(acc, off, 32);
  if (lane == 0) {
    float gv = acc + bg2[0];
    gate[node] = gv;
    atomicMaxF(&gmax[batch[node]], gv);
  }
}

__global__ void gate_sum_kernel(const float* __restrict__ gate, const int* __restrict__ batch,
                                const float* __restrict__ gmax,
                                float* __restrict__ gp, float* __restrict__ gs, int Nn)
{
  int n = blockIdx.x * blockDim.x + threadIdx.x;
  if (n >= Nn) return;
  int bb = batch[n];
  float p = __expf(gate[n] - gmax[bb]);
  gp[n] = p;
  atomicAdd(&gs[bb], p);
}

__global__ void pool_kernel(const float* __restrict__ h, const float* __restrict__ gp,
                            const float* __restrict__ gs, const int* __restrict__ batch,
                            float* __restrict__ out, int Nn)
{
  long i = (long)blockIdx.x * blockDim.x + threadIdx.x;
  if (i >= (long)Nn * HIDC) return;
  int n = (int)(i >> 7), c = (int)(i & 127);
  int bb = batch[n];
  float w = gp[n] / (gs[bb] + 1e-16f);
  atomicAdd(&out[(size_t)bb * HIDC + c], w * h[(size_t)n * HIDC + c]);
}

// ---------------------------------------------------------------------------
// host launcher
// ---------------------------------------------------------------------------
extern "C" void kernel_launch(void* const* d_in, const int* in_sizes, int n_in,
                              void* d_out, int out_size, void* d_ws, size_t ws_size,
                              hipStream_t stream) {
  (void)n_in; (void)ws_size;
  const float* x     = (const float*)d_in[0];
  const float* ea    = (const float*)d_in[1];
  const float* Win   = (const float*)d_in[2];
  const float* b_in  = (const float*)d_in[3];
  const float* We1   = (const float*)d_in[4];
  const float* be1   = (const float*)d_in[5];
  const float* We2   = (const float*)d_in[6];
  const float* be2   = (const float*)d_in[7];
  const float* Wl    = (const float*)d_in[8];
  const float* a_src = (const float*)d_in[9];
  const float* a_dst = (const float*)d_in[10];
  const float* bl    = (const float*)d_in[11];
  const float* ln_g  = (const float*)d_in[12];
  const float* ln_b  = (const float*)d_in[13];
  const float* Wg1   = (const float*)d_in[14];
  const float* bg1   = (const float*)d_in[15];
  const float* Wg2   = (const float*)d_in[16];
  const float* bg2   = (const float*)d_in[17];
  const int*   ei    = (const int*)d_in[18];
  const int*   batch = (const int*)d_in[19];

  const int N = in_sizes[19];
  const int E = in_sizes[18] / 2;
  const int B = out_size / HIDC;
  const int* srcs = ei;
  const int* dsts = ei + E;

  float* ws   = (float*)d_ws;
  float* h    = ws;  ws += (size_t)N * HIDC;
  float* xp   = ws;  ws += (size_t)N * HIDC;
  float* agg  = ws;  ws += (size_t)N * HIDC;
  float* as_n = ws;  ws += (size_t)N * HEADS;
  float* ad_n = ws;  ws += (size_t)N * HEADS;
  float* mbuf = ws;  ws += (size_t)N * HEADS;
  float* sbuf = ws;  ws += (size_t)N * HEADS;
  float* crit = ws;  ws += N;
  float* gatev= ws;  ws += N;
  float* gpv  = ws;  ws += N;
  float* gmax = ws;  ws += B;
  float* gs   = ws;  ws += B;
  // packed f16 weights (5 matrices), 16B aligned for uint4 staging
  uintptr_t wp_addr = ((uintptr_t)ws + 15) & ~(uintptr_t)15;
  _Float16* wpack = (_Float16*)wp_addr;      // 5 * 16384 halves = 160KB
  float* outf = (float*)d_out;

  const float NEG_INF = -__builtin_inff();
  auto cdiv = [](long a, long b) { return (int)((a + b - 1) / b); };
  const int mtiles = cdiv(N, 64);

  // 0) pack all 5 weight matrices into WMMA B-fragment order
  pack_weight_kernel<<<WPACK_ELEMS / 256, 256, 0, stream>>>(Win, wpack);
  for (int l = 0; l < NLAYER; ++l)
    pack_weight_kernel<<<WPACK_ELEMS / 256, 256, 0, stream>>>(
        Wl + (size_t)l * HIDC * HIDC, wpack + (size_t)(l + 1) * WPACK_ELEMS);

  // 1) input projection
  gemm128_wmma_kernel<<<mtiles, 256, 0, stream>>>(x, wpack, b_in, h, N);

  // 2) crit signal from edge MLP
  fill_f32<<<cdiv(N, 256), 256, 0, stream>>>(crit, 0.f, N);
  edge_mlp_kernel<<<cdiv(E, 256), 256, 0, stream>>>(ea, We1, be1, We2, be2, dsts, crit, E);

  // 3) GAT layers
  for (int l = 0; l < NLAYER; ++l) {
    gemm128_wmma_kernel<<<mtiles, 256, 0, stream>>>(
        h, wpack + (size_t)(l + 1) * WPACK_ELEMS, nullptr, xp, N);
    attn_coef_kernel<<<cdiv((long)N * HEADS, 256), 256, 0, stream>>>(
        xp, a_src + (size_t)l * HEADS * CHPH, a_dst + (size_t)l * HEADS * CHPH,
        as_n, ad_n, N);
    fill_f32<<<cdiv((long)N * HEADS, 256), 256, 0, stream>>>(mbuf, NEG_INF, (long)N * HEADS);
    fill_f32<<<cdiv((long)N * HEADS, 256), 256, 0, stream>>>(sbuf, 0.f, (long)N * HEADS);
    fill_f32<<<cdiv((long)N * HIDC, 256), 256, 0, stream>>>(agg, 0.f, (long)N * HIDC);
    edge_max_kernel<<<cdiv((long)(E + N) * HEADS, 256), 256, 0, stream>>>(
        as_n, ad_n, srcs, dsts, mbuf, E, N);
    edge_sum_kernel<<<cdiv((long)(E + N) * HEADS, 256), 256, 0, stream>>>(
        as_n, ad_n, srcs, dsts, mbuf, sbuf, E, N);
    aggregate_kernel<<<cdiv((long)(E + N) * 32, 256), 256, 0, stream>>>(
        xp, as_n, ad_n, mbuf, sbuf, srcs, dsts, agg, E, N);
    post_kernel<<<N, 128, 0, stream>>>(
        agg, bl + (size_t)l * HIDC, ln_g + (size_t)l * HIDC, ln_b + (size_t)l * HIDC,
        crit, h, N);
  }

  // 4) global-attention pooling
  fill_f32<<<1, 256, 0, stream>>>(gmax, NEG_INF, B);
  fill_f32<<<1, 256, 0, stream>>>(gs, 0.f, B);
  fill_f32<<<cdiv(out_size, 256), 256, 0, stream>>>(outf, 0.f, out_size);
  gate_kernel<<<cdiv((long)N * 32, 256), 256, 0, stream>>>(
      h, Wg1, bg1, Wg2, bg2, batch, gatev, gmax, N);
  gate_sum_kernel<<<cdiv(N, 256), 256, 0, stream>>>(gatev, batch, gmax, gpv, gs, N);
  pool_kernel<<<cdiv((long)N * HIDC, 256), 256, 0, stream>>>(h, gpv, gs, batch, outf, N);
}
